// SoAGREE_47021301956985
// MI455X (gfx1250) — matrix operand order repaired and checked by reading, not compiled, not run on hardware
//
#include <hip/hip_runtime.h>
#include <hip/hip_bf16.h>

// SoAGREE on MI455X (gfx1250, wave32). Batched matmuls via
// V_WMMA_F32_16X16X32_F16 (f32 accumulate); gathers via
// GLOBAL_LOAD_ASYNC_TO_LDS_B128 (ASYNCcnt) + s_wait_asynccnt.

typedef __attribute__((ext_vector_type(16))) _Float16 v16h;
typedef __attribute__((ext_vector_type(8)))  float    v8f;

#define DD   64   // embedding dim
#define HH   64   // hidden dim
#define KNB  32   // padded neighbors per user
#define MMEM 16   // padded members per team

static __device__ __forceinline__ v8f wmma_f16(v16h a, v16h b, v8f c) {
    // (neg_a, A, neg_b, B, c_mod, C, reuse_a, reuse_b)
    return __builtin_amdgcn_wmma_f32_16x16x32_f16(false, a, false, b, (short)0, c,
                                                  false, false);
}

// Async copy of 16 bytes per lane: global -> LDS, tracked by ASYNCcnt.
// vdst = 32-bit LDS byte address (low 32 bits of the flat pointer),
// vaddr = 64-bit global address, saddr = off.
static __device__ __forceinline__ void async_copy_b128(const float* gsrc, float* ldst) {
    unsigned lds_addr = (unsigned)(size_t)ldst;
    asm volatile("global_load_async_to_lds_b128 %0, %1, off"
                 :: "v"(lds_addr), "v"(gsrc)
                 : "memory");
}

static __device__ __forceinline__ void async_wait0() {
    asm volatile("s_wait_asynccnt 0x0" ::: "memory");
}

// ---------------------------------------------------------------------------
// Kernel 0: repo-side constants shared by all teams.
//   repoW[h] = ba1[h] + sum_d repo[d]*Wa1[D+d][h]
//   rpW[h]   = bp1[h] + sum_d repo[d]*Wp1[2D+d][h]
// ---------------------------------------------------------------------------
__global__ void consts_kernel(const float* __restrict__ repo,
                              const float* __restrict__ Wa1,
                              const float* __restrict__ ba1,
                              const float* __restrict__ Wp1,
                              const float* __restrict__ bp1,
                              float* __restrict__ repoW,
                              float* __restrict__ rpW) {
    int h = threadIdx.x;               // 64 threads
    float a = ba1[h], p = bp1[h];
    for (int d = 0; d < DD; ++d) {
        float r = repo[d];
        a += r * Wa1[(DD + d) * HH + h];
        p += r * Wp1[(2 * DD + d) * HH + h];
    }
    repoW[h] = a;
    rpW[h]   = p;
}

// ---------------------------------------------------------------------------
// Kernel 1: user aggregation. One wave (32 lanes) per user, 4 waves/block.
// ---------------------------------------------------------------------------
__global__ void __launch_bounds__(128)
user_agg_kernel(const float* __restrict__ user_embeds,
                const float* __restrict__ Wf1,
                const float* __restrict__ bf1,
                const float* __restrict__ Wf2,
                const int*   __restrict__ user_neighbors,
                const int*   __restrict__ neighbor_counts,
                float* __restrict__ user_agg, int U) {
    __shared__ v16h  sB[2][4][32];        // Wf1[:D] f16 B-fragments [kt][nt][lane]
    __shared__ float sWf2[HH];
    __shared__ float sNeigh[4][KNB][DD];  // per-wave gathered neighbor rows (f32)
    __shared__ float sSelf[4][HH];        // per-wave self term (incl. bf1)

    const int tid  = threadIdx.x;
    const int wave = tid >> 5, lane = tid & 31;
    const int half = lane >> 4, lcol = lane & 15;

    // Block-level preload: convert Wf1[:D] into per-lane WMMA B fragments.
    for (int idx = tid; idx < 2 * 4 * 32; idx += 128) {
        int ln = idx & 31, nt = (idx >> 5) & 3, kt = idx >> 7;
        int col = nt * 16 + (ln & 15);
        v16h t;
#pragma unroll
        for (int i = 0; i < 16; ++i) {
            int Kk = ((ln >> 4) * 16) + i;            // B: lanes 0-15 K=0..15, 16-31 K=16..31
            t[i] = (_Float16)Wf1[(kt * 32 + Kk) * HH + col];
        }
        sB[kt][nt][ln] = t;
    }
    if (tid < HH) sWf2[tid] = Wf2[tid];
    __syncthreads();

    const int u = blockIdx.x * 4 + wave;
    if (u >= U) return;

    const int cnt  = neighbor_counts[u];
    const int nidx = user_neighbors[u * KNB + lane];

    // Async gather: 32 neighbor rows (256 B each) into LDS.
    // Each wave-wide b128 op moves two rows (lanes 0-15 row 2rp, 16-31 row 2rp+1).
#pragma unroll
    for (int rp = 0; rp < KNB / 2; ++rp) {
        int row = rp * 2 + half;
        int j   = __shfl(nidx, row, 32);
        const float* gsrc = user_embeds + (size_t)j * DD + lcol * 4;
        async_copy_b128(gsrc, &sNeigh[wave][row][lcol * 4]);
    }

    // Self term while the async gather is in flight:
    // st[h] = bf1[h] + sum_d self[d]*Wf1[D+d][h]  (64x64 matvec)
    float self0 = user_embeds[(size_t)u * DD + lane];
    float self1 = user_embeds[(size_t)u * DD + lane + 32];
    {
        float st0 = bf1[lane], st1 = bf1[lane + 32];
        for (int d = 0; d < DD; ++d) {
            float sd = (d < 32) ? __shfl(self0, d, 32) : __shfl(self1, d - 32, 32);
            st0 += sd * Wf1[(DD + d) * HH + lane];
            st1 += sd * Wf1[(DD + d) * HH + lane + 32];
        }
        sSelf[wave][lane]      = st0;
        sSelf[wave][lane + 32] = st1;
    }
    async_wait0();                     // gather complete
    __builtin_amdgcn_wave_barrier();   // same-wave LDS ops are in-order on CDNA5

    // WMMA: [32,64] x [64,64] -> H in 2x4 f32 accumulator tiles.
    v8f z = {};
    v8f c[2][4];
#pragma unroll
    for (int mt = 0; mt < 2; ++mt)
#pragma unroll
        for (int nt = 0; nt < 4; ++nt) c[mt][nt] = z;

#pragma unroll
    for (int kt = 0; kt < 2; ++kt) {
        v16h a[2];
#pragma unroll
        for (int mt = 0; mt < 2; ++mt) {
            int row = mt * 16 + lcol;
            v16h t;
#pragma unroll
            for (int i = 0; i < 16; ++i) {
                int Kk = half * 16 + i;   // A: contiguous K order, matches B fragments
                t[i] = (_Float16)sNeigh[wave][row][kt * 32 + Kk];
            }
            a[mt] = t;
        }
#pragma unroll
        for (int nt = 0; nt < 4; ++nt) {
            v16h b = sB[kt][nt][lane];
            c[0][nt] = wmma_f16(a[0], b, c[0][nt]);
            c[1][nt] = wmma_f16(a[1], b, c[1][nt]);
        }
    }

    // Scores s_k = sum_h relu(H[k][h]+st[h]) * Wf2[h]; reduce across 16 lanes.
    float prow[2][8];
#pragma unroll
    for (int mt = 0; mt < 2; ++mt)
#pragma unroll
        for (int r = 0; r < 8; ++r) prow[mt][r] = 0.f;
#pragma unroll
    for (int nt = 0; nt < 4; ++nt) {
        int colh = nt * 16 + lcol;
        float stc = sSelf[wave][colh];
        float w2  = sWf2[colh];
#pragma unroll
        for (int mt = 0; mt < 2; ++mt)
#pragma unroll
            for (int r = 0; r < 8; ++r) {
                float hv = fmaxf(c[mt][nt][r] + stc, 0.f);
                prow[mt][r] += hv * w2;
            }
    }
#pragma unroll
    for (int m = 1; m <= 8; m <<= 1)
#pragma unroll
        for (int mt = 0; mt < 2; ++mt)
#pragma unroll
            for (int r = 0; r < 8; ++r)
                prow[mt][r] += __shfl_xor(prow[mt][r], m, 32);
    // Lane now holds s for rows {mt*16 + half*8 + r}. Softmax fully in registers.
    float mv = -3.0e38f;
#pragma unroll
    for (int mt = 0; mt < 2; ++mt)
#pragma unroll
        for (int r = 0; r < 8; ++r) {
            bool v = (mt * 16 + half * 8 + r) < cnt;
            mv = v ? fmaxf(mv, prow[mt][r]) : mv;
        }
    mv = fmaxf(mv, __shfl_xor(mv, 16, 32));
    float w_[2][8];
    float es = 0.f;
#pragma unroll
    for (int mt = 0; mt < 2; ++mt)
#pragma unroll
        for (int r = 0; r < 8; ++r) {
            bool v = (mt * 16 + half * 8 + r) < cnt;
            float e = v ? __expf(prow[mt][r] - mv) : 0.f;
            w_[mt][r] = e;
            es += e;
        }
    es += __shfl_xor(es, 16, 32);
    float inv = (es > 0.f) ? 1.f / es : 0.f;
#pragma unroll
    for (int mt = 0; mt < 2; ++mt)
#pragma unroll
        for (int r = 0; r < 8; ++r) w_[mt][r] *= inv;

    // agg[d] = sum_k w_k * neigh[k][d]; lane pair (l, l^16) covers 4 columns.
    float part[4] = {0.f, 0.f, 0.f, 0.f};
#pragma unroll
    for (int mt = 0; mt < 2; ++mt)
#pragma unroll
        for (int r = 0; r < 8; ++r) {
            int row = mt * 16 + half * 8 + r;
            float wk = w_[mt][r];
#pragma unroll
            for (int j = 0; j < 4; ++j)
                part[j] += wk * sNeigh[wave][row][lcol * 4 + j];
        }
#pragma unroll
    for (int j = 0; j < 4; ++j) part[j] += __shfl_xor(part[j], 16, 32);
    float p0 = half ? part[2] : part[0];
    float p1 = half ? part[3] : part[1];
    int   c0 = lcol * 4 + half * 2;
    float s0 = user_embeds[(size_t)u * DD + c0];
    float s1 = user_embeds[(size_t)u * DD + c0 + 1];
    user_agg[(size_t)u * DD + c0]     = (cnt > 0) ? p0 + s0 : s0;
    user_agg[(size_t)u * DD + c0 + 1] = (cnt > 0) ? p1 + s1 : s1;
}

// ---------------------------------------------------------------------------
// Kernel 2: group aggregation. One wave per team.
// ---------------------------------------------------------------------------
__global__ void __launch_bounds__(128)
group_agg_kernel(const float* __restrict__ user_agg,
                 const float* __restrict__ team_embeds,
                 const float* __restrict__ Wa1,
                 const float* __restrict__ Wa2,
                 const float* __restrict__ repoW,
                 const int*   __restrict__ team_members,
                 const int*   __restrict__ member_counts,
                 float* __restrict__ group_out, int G) {
    __shared__ v16h  sB[2][4][32];
    __shared__ float sWa2[HH];
    __shared__ float sRW[HH];
    __shared__ float sMem[4][MMEM][DD];

    const int tid  = threadIdx.x;
    const int wave = tid >> 5, lane = tid & 31;
    const int half = lane >> 4, lcol = lane & 15;

    for (int idx = tid; idx < 2 * 4 * 32; idx += 128) {
        int ln = idx & 31, nt = (idx >> 5) & 3, kt = idx >> 7;
        int col = nt * 16 + (ln & 15);
        v16h t;
#pragma unroll
        for (int i = 0; i < 16; ++i) {
            int Kk = ((ln >> 4) * 16) + i;
            t[i] = (_Float16)Wa1[(kt * 32 + Kk) * HH + col];
        }
        sB[kt][nt][ln] = t;
    }
    if (tid < HH) { sWa2[tid] = Wa2[tid]; sRW[tid] = repoW[tid]; }
    __syncthreads();

    const int g = blockIdx.x * 4 + wave;
    if (g >= G) return;

    const int cnt  = member_counts[g];
    const int midx = team_members[g * MMEM + lcol];

    // Async gather: 16 member rows, two rows per wave-wide b128 op.
#pragma unroll
    for (int rp = 0; rp < MMEM / 2; ++rp) {
        int row = rp * 2 + half;
        int j   = __shfl(midx, row, 32);
        const float* gsrc = user_agg + (size_t)j * DD + lcol * 4;
        async_copy_b128(gsrc, &sMem[wave][row][lcol * 4]);
    }
    async_wait0();
    __builtin_amdgcn_wave_barrier();

    v8f z = {};
    v8f c[4] = {z, z, z, z};
#pragma unroll
    for (int kt = 0; kt < 2; ++kt) {
        v16h a;
#pragma unroll
        for (int i = 0; i < 16; ++i) {
            int Kk = half * 16 + i;
            a[i] = (_Float16)sMem[wave][lcol][kt * 32 + Kk];
        }
#pragma unroll
        for (int nt = 0; nt < 4; ++nt)
            c[nt] = wmma_f16(a, sB[kt][nt][lane], c[nt]);
    }

    float prow[8];
#pragma unroll
    for (int r = 0; r < 8; ++r) prow[r] = 0.f;
#pragma unroll
    for (int nt = 0; nt < 4; ++nt) {
        int colh = nt * 16 + lcol;
        float rw = sRW[colh];
        float w2 = sWa2[colh];
#pragma unroll
        for (int r = 0; r < 8; ++r) {
            float hv = fmaxf(c[nt][r] + rw, 0.f);
            prow[r] += hv * w2;
        }
    }
#pragma unroll
    for (int m = 1; m <= 8; m <<= 1)
#pragma unroll
        for (int r = 0; r < 8; ++r)
            prow[r] += __shfl_xor(prow[r], m, 32);

    float mv = -3.0e38f;
#pragma unroll
    for (int r = 0; r < 8; ++r) {
        bool v = (half * 8 + r) < cnt;
        mv = v ? fmaxf(mv, prow[r]) : mv;
    }
    mv = fmaxf(mv, __shfl_xor(mv, 16, 32));
    float e[8];
    float es = 0.f;
#pragma unroll
    for (int r = 0; r < 8; ++r) {
        bool v = (half * 8 + r) < cnt;
        e[r] = v ? __expf(prow[r] - mv) : 0.f;
        es += e[r];
    }
    es += __shfl_xor(es, 16, 32);
    float inv = (es > 0.f) ? 1.f / es : 0.f;

    float part[4] = {0.f, 0.f, 0.f, 0.f};
#pragma unroll
    for (int r = 0; r < 8; ++r) {
        int row = half * 8 + r;
        float wk = e[r] * inv;
#pragma unroll
        for (int j = 0; j < 4; ++j)
            part[j] += wk * sMem[wave][row][lcol * 4 + j];
    }
#pragma unroll
    for (int j = 0; j < 4; ++j) part[j] += __shfl_xor(part[j], 16, 32);
    float p0 = half ? part[2] : part[0];
    float p1 = half ? part[3] : part[1];
    int   c0 = lcol * 4 + half * 2;
    float t0 = team_embeds[(size_t)g * DD + c0];
    float t1 = team_embeds[(size_t)g * DD + c0 + 1];
    group_out[(size_t)g * DD + c0]     = (cnt > 0) ? p0 + t0 : t0;
    group_out[(size_t)g * DD + c0 + 1] = (cnt > 0) ? p1 + t1 : t1;
}

// ---------------------------------------------------------------------------
// Kernel 3: predict. One wave handles 16 teams via two fused WMMA GEMMs.
// ---------------------------------------------------------------------------
__global__ void __launch_bounds__(128)
predict_kernel(const float* __restrict__ group,
               const float* __restrict__ repo,
               const float* __restrict__ Wp1,
               const float* __restrict__ Wp2,
               const float* __restrict__ bp2p,
               const float* __restrict__ rpW,
               float* __restrict__ out, int G) {
    __shared__ v16h  sB[2][2][4][32];   // [which(Wp1a/Wp1b)][kt][nt][lane]
    __shared__ float sRepo[DD];
    __shared__ float sWp2[HH];
    __shared__ float sRW[HH];

    const int tid  = threadIdx.x;
    const int wave = tid >> 5, lane = tid & 31;
    const int half = lane >> 4, lcol = lane & 15;

    for (int idx = tid; idx < 2 * 2 * 4 * 32; idx += 128) {
        int ln = idx & 31, nt = (idx >> 5) & 3, kt = (idx >> 7) & 1, w = idx >> 8;
        int col = nt * 16 + (ln & 15);
        v16h t;
#pragma unroll
        for (int i = 0; i < 16; ++i) {
            int Kk = ((ln >> 4) * 16) + i;
            t[i] = (_Float16)Wp1[(w * DD + kt * 32 + Kk) * HH + col];
        }
        sB[w][kt][nt][ln] = t;
    }
    if (tid < DD) sRepo[tid] = repo[tid];
    if (tid < HH) { sWp2[tid] = Wp2[tid]; sRW[tid] = rpW[tid]; }
    __syncthreads();

    const int base = (blockIdx.x * 4 + wave) * 16;
    if (base >= G) return;
    const float bp2v = bp2p[0];

    int t_ = base + lcol;
    int tt = (t_ < G) ? t_ : (G - 1);
    const float* grow = group + (size_t)tt * DD;

    v8f z = {};
    v8f c[4] = {z, z, z, z};
#pragma unroll
    for (int kt = 0; kt < 2; ++kt) {
        float gv[16];
#pragma unroll
        for (int c2 = 0; c2 < 2; ++c2) {
            int bc = kt * 32 + half * 16 + c2 * 8;
            float4 f0 = *(const float4*)(grow + bc);
            float4 f1 = *(const float4*)(grow + bc + 4);
            gv[c2 * 8 + 0] = f0.x; gv[c2 * 8 + 1] = f0.y;
            gv[c2 * 8 + 2] = f0.z; gv[c2 * 8 + 3] = f0.w;
            gv[c2 * 8 + 4] = f1.x; gv[c2 * 8 + 5] = f1.y;
            gv[c2 * 8 + 6] = f1.z; gv[c2 * 8 + 7] = f1.w;
        }
        v16h a1, a2;
#pragma unroll
        for (int i = 0; i < 16; ++i) {
            int Kk  = half * 16 + i;
            int col = kt * 32 + Kk;
            float g = gv[i];
            a2[i] = (_Float16)g;                  // group row
            a1[i] = (_Float16)(g * sRepo[col]);   // group*repo row
        }
#pragma unroll
        for (int nt = 0; nt < 4; ++nt) {
            c[nt] = wmma_f16(a1, sB[0][kt][nt][lane], c[nt]);
            c[nt] = wmma_f16(a2, sB[1][kt][nt][lane], c[nt]);
        }
    }

    float prow[8];
#pragma unroll
    for (int r = 0; r < 8; ++r) prow[r] = 0.f;
#pragma unroll
    for (int nt = 0; nt < 4; ++nt) {
        int colh = nt * 16 + lcol;
        float rw = sRW[colh];
        float w2 = sWp2[colh];
#pragma unroll
        for (int r = 0; r < 8; ++r) {
            float hv = fmaxf(c[nt][r] + rw, 0.f);
            prow[r] += hv * w2;
        }
    }
#pragma unroll
    for (int m = 1; m <= 8; m <<= 1)
#pragma unroll
        for (int r = 0; r < 8; ++r)
            prow[r] += __shfl_xor(prow[r], m, 32);

#pragma unroll
    for (int r = 0; r < 8; ++r) {
        if (lcol == r) {
            int t2 = base + half * 8 + r;
            if (t2 < G) {
                float s = prow[r] + bp2v;
                out[t2] = 1.f / (1.f + __expf(-s));
            }
        }
    }
}

// ---------------------------------------------------------------------------
extern "C" void kernel_launch(void* const* d_in, const int* in_sizes, int n_in,
                              void* d_out, int out_size, void* d_ws, size_t ws_size,
                              hipStream_t stream) {
    (void)n_in; (void)out_size; (void)ws_size;
    const float* repo            = (const float*)d_in[0];
    const float* team_embeds     = (const float*)d_in[1];
    const float* user_embeds     = (const float*)d_in[2];
    const float* Wf1             = (const float*)d_in[3];
    const float* bf1             = (const float*)d_in[4];
    const float* Wf2             = (const float*)d_in[5];
    // d_in[6] = bf2 (uniform shift; cancels in softmax)
    const float* Wa1             = (const float*)d_in[7];
    const float* ba1             = (const float*)d_in[8];
    const float* Wa2             = (const float*)d_in[9];
    // d_in[10] = ba2 (uniform shift; cancels in softmax)
    const float* Wp1             = (const float*)d_in[11];
    const float* bp1             = (const float*)d_in[12];
    const float* Wp2             = (const float*)d_in[13];
    const float* bp2             = (const float*)d_in[14];
    const int*   team_members    = (const int*)d_in[15];
    const int*   member_counts   = (const int*)d_in[16];
    const int*   user_neighbors  = (const int*)d_in[17];
    const int*   neighbor_counts = (const int*)d_in[18];

    const int U = in_sizes[2] / DD;   // 50000
    const int G = in_sizes[1] / DD;   // 20000

    float* user_agg = (float*)d_ws;
    float* group    = user_agg + (size_t)U * DD;
    float* repoW    = group + (size_t)G * DD;
    float* rpW      = repoW + HH;

    consts_kernel<<<1, HH, 0, stream>>>(repo, Wa1, ba1, Wp1, bp1, repoW, rpW);
    user_agg_kernel<<<(U + 3) / 4, 128, 0, stream>>>(
        user_embeds, Wf1, bf1, Wf2, user_neighbors, neighbor_counts, user_agg, U);
    group_agg_kernel<<<(G + 3) / 4, 128, 0, stream>>>(
        user_agg, team_embeds, Wa1, Wa2, repoW, team_members, member_counts, group, G);
    predict_kernel<<<(G + 63) / 64, 128, 0, stream>>>(
        group, repo, Wp1, Wp2, bp2, rpW, (float*)d_out, G);
}
